// Forward_Multihead_Attention_light_70626442215645
// MI455X (gfx1250) — compile-verified, matrix operand order checked by reading, not run on hardware
//
#include <hip/hip_runtime.h>
#include <math.h>

// ---------------------------------------------------------------------------
// MI455X (gfx1250) implementation.
//  * BN folded into weights (eval-mode BN is affine) -> 6 pure GEMM+act passes
//  * GEMMs use v_wmma_f32_16x16x32_f16 (f16 in, f32 accumulate)
//  * attention + sparsemax: one block per query row, one lane per key,
//    bitonic sort + scan in LDS (k_dict f16 = 16.7MB -> L2 resident)
//
// Input flattening assumption (top level = dict order, nested params = JAX
// pytree sorted-key order):
//   d_in[0]=x(2048*1*512 f32)  d_in[1]=scene(2048 i32)  d_in[2]=anch(64*256*512 f32)
//   params: "bn" < "wk" < "wq"
//     bn  : beta(8) gamma(8) rm(8) rv(8)              -> d_in[3..6]
//     wk  : 3 layers x sorted{W,b,beta,gamma,rm,rv}   -> d_in[7..24]
//     wq  : 3 layers x sorted{W,b,beta,gamma,rm,rv}   -> d_in[25..42]
// ---------------------------------------------------------------------------

typedef __attribute__((ext_vector_type(16))) _Float16 v16h;
typedef __attribute__((ext_vector_type(8)))  float    v8f;
typedef _Float16 f16;

#define BN_EPS 1e-5f

// ---- fold eval-mode BatchNorm into a Linear layer's weights/bias ----------
__global__ void fold_bn_kernel(const float* __restrict__ W, const float* __restrict__ b,
                               const float* __restrict__ gamma, const float* __restrict__ beta,
                               const float* __restrict__ rm, const float* __restrict__ rv,
                               f16* __restrict__ Wf, float* __restrict__ bf) {
    const int n = blockIdx.x;                         // output feature (512)
    const float s = gamma[n] * rsqrtf(rv[n] + BN_EPS);
    const int base = n * 512;
    for (int c = threadIdx.x; c < 512; c += blockDim.x)
        Wf[base + c] = (f16)(W[base + c] * s);
    if (threadIdx.x == 0)
        bf[n] = (b[n] - rm[n]) * s + beta[n];
}

// ---- per-channel head-BN scale/shift (512 channels, 8 heads x 64) ---------
__global__ void head_bn_kernel(const float* __restrict__ g, const float* __restrict__ be,
                               const float* __restrict__ rm, const float* __restrict__ rv,
                               float* __restrict__ hscale, float* __restrict__ hshift) {
    const int c = threadIdx.x;                        // 512
    const int h = c >> 6;
    const float s = g[h] * rsqrtf(rv[h] + BN_EPS);
    hscale[c] = s;
    hshift[c] = be[h] - rm[h] * s;
}

// ---- f32 -> f16 conversion ------------------------------------------------
__global__ void cvt_f32_f16(const float* __restrict__ in, f16* __restrict__ out, int n) {
    const int i = blockIdx.x * blockDim.x + threadIdx.x;
    if (i < n) out[i] = (f16)in[i];
}

// ---- WMMA GEMM:  Y = act(X[Mx512] @ W[512x512]^T + bias)  -----------------
// MODE 0: ReLU            (layers 1,2)
// MODE 1: sigmoid -> head-BN affine  (layer 3)
template <int MODE>
__global__ __launch_bounds__(256) void gemm_bn_act(
    const f16* __restrict__ X,        // M x 512 row-major (f16)
    const f16* __restrict__ W,        // 512 x 512 row-major, rows = out features
    const float* __restrict__ bias,   // 512 (BN-folded)
    const float* __restrict__ hscale, // 512 (MODE 1)
    const float* __restrict__ hshift, // 512 (MODE 1)
    f16* __restrict__ Y)              // M x 512
{
    // LDS tiles, 40-half row stride (80B) -> conflict-free b128 fragment reads
    __shared__ __align__(16) f16 sA[128][40];
    __shared__ __align__(16) f16 sB[128][40];

    const int tid  = threadIdx.x;
    const int lane = tid & 31;
    const int wave = tid >> 5;        // 8 waves
    const int wm   = wave & 3;        // 32-row group within 128
    const int wn   = wave >> 2;       // 64-col group within 128
    const int m0   = blockIdx.x * 128;
    const int n0   = blockIdx.y * 128;

    v8f acc[2][4];
    const v8f vzero = {0.f, 0.f, 0.f, 0.f, 0.f, 0.f, 0.f, 0.f};
    for (int i = 0; i < 2; ++i)
        for (int j = 0; j < 4; ++j) acc[i][j] = vzero;

    // cooperative tile load mapping: 2 threads per row, 16 halves each
    const int lr = tid >> 1;
    const int lc = (tid & 1) * 16;

    const int mrow   = lane & 15;
    const int kchunk = (lane >> 4) * 8;   // ISA 16-bit A/B layout: K = kchunk..+7, kchunk+16..+23

    for (int kc = 0; kc < 512; kc += 32) {
        const uint4* gA = (const uint4*)(X + (size_t)(m0 + lr) * 512 + kc + lc);
        const uint4* gB = (const uint4*)(W + (size_t)(n0 + lr) * 512 + kc + lc);
        uint4 a0 = gA[0], a1 = gA[1];
        uint4 b0 = gB[0], b1 = gB[1];
        *(uint4*)&sA[lr][lc]     = a0;
        *(uint4*)&sA[lr][lc + 8] = a1;
        *(uint4*)&sB[lr][lc]     = b0;
        *(uint4*)&sB[lr][lc + 8] = b1;
        if (kc + 32 < 512) {   // gfx1250 global_prefetch_b8 on next K-chunk
            __builtin_prefetch((const void*)(X + (size_t)(m0 + lr) * 512 + kc + 32 + lc), 0, 1);
            __builtin_prefetch((const void*)(W + (size_t)(n0 + lr) * 512 + kc + 32 + lc), 0, 1);
        }
        __syncthreads();

        union Frag { v16h h; uint4 u[2]; };
        Frag fa[2], fb[4];
        #pragma unroll
        for (int mt = 0; mt < 2; ++mt) {
            const int r = wm * 32 + mt * 16 + mrow;
            fa[mt].u[0] = *(const uint4*)&sA[r][kchunk];
            fa[mt].u[1] = *(const uint4*)&sA[r][kchunk + 16];
        }
        #pragma unroll
        for (int nt = 0; nt < 4; ++nt) {
            const int r = wn * 64 + nt * 16 + mrow;
            fb[nt].u[0] = *(const uint4*)&sB[r][kchunk];
            fb[nt].u[1] = *(const uint4*)&sB[r][kchunk + 16];
        }
        #pragma unroll
        for (int mt = 0; mt < 2; ++mt)
            #pragma unroll
            for (int nt = 0; nt < 4; ++nt)
                acc[mt][nt] = __builtin_amdgcn_wmma_f32_16x16x32_f16(
                    false, fa[mt].h, false, fb[nt].h,
                    (short)0, acc[mt][nt], false, false);
        __syncthreads();
    }

    // epilogue: C/D layout -> lane = N%16, VGPR r = M (lanes>=16: M=r+8)
    const int nlane = lane & 15;
    const int rowhi = (lane >> 4) * 8;
    #pragma unroll
    for (int mt = 0; mt < 2; ++mt)
        #pragma unroll
        for (int nt = 0; nt < 4; ++nt) {
            const int col = n0 + wn * 64 + nt * 16 + nlane;
            const float bv = bias[col];
            float hs = 0.f, hb = 0.f;
            if (MODE == 1) { hs = hscale[col]; hb = hshift[col]; }
            #pragma unroll
            for (int r = 0; r < 8; ++r) {
                const int row = m0 + wm * 32 + mt * 16 + rowhi + r;
                float v = acc[mt][nt][r] + bv;
                if (MODE == 0) {
                    v = fmaxf(v, 0.0f);
                } else {
                    v = 1.0f / (1.0f + __expf(-v));   // sigmoid
                    v = v * hs + hb;                   // head BN
                }
                Y[(size_t)row * 512 + col] = (f16)v;
            }
        }
}

// ---- gathered attention + sparsemax: 1 block per query, 1 lane per key ----
__global__ __launch_bounds__(256) void attn_sparsemax(
    const f16* __restrict__ Qb,     // 2048 x 512
    const f16* __restrict__ Kb,     // 16384 x 512 (k_dict, L2 resident)
    const int* __restrict__ scene,  // 2048
    float* __restrict__ out)        // 2048 x 256
{
    __shared__ __align__(16) f16 qs[512];
    __shared__ float srt[256];
    __shared__ float cs[256];
    __shared__ int   cnt;

    const int b = blockIdx.x;
    const int t = threadIdx.x;
    if (t < 64) ((uint4*)qs)[t] = ((const uint4*)(Qb + (size_t)b * 512))[t];
    if (t == 0) cnt = 0;
    __syncthreads();

    const int s = scene[b];
    const f16* Kr = Kb + ((size_t)s * 256 + t) * 512;

    // max over 8 heads of 64-dim dots; final scale 1/(sqrt(HD)*sqrt(HD)) = 1/64
    float best = -3.4e38f;
    #pragma unroll
    for (int h = 0; h < 8; ++h) {
        float acc = 0.0f;
        const uint4* kq = (const uint4*)(Kr + h * 64);
        const uint4* qq = (const uint4*)(qs + h * 64);
        #pragma unroll
        for (int c = 0; c < 8; ++c) {
            uint4 kv = kq[c];
            uint4 qv = qq[c];
            const f16* kh = (const f16*)&kv;
            const f16* qh = (const f16*)&qv;
            #pragma unroll
            for (int e = 0; e < 8; ++e)
                acc += (float)qh[e] * (float)kh[e];
        }
        best = fmaxf(best, acc);
    }
    const float z = best * (1.0f / 64.0f);
    srt[t] = z;
    __syncthreads();

    // bitonic sort, descending, 256 elements
    for (int k = 2; k <= 256; k <<= 1) {
        for (int j = k >> 1; j > 0; j >>= 1) {
            const int ixj = t ^ j;
            if (ixj > t) {
                const float a = srt[t], c2 = srt[ixj];
                const bool desc = ((t & k) == 0);
                if (desc ? (a < c2) : (a > c2)) { srt[t] = c2; srt[ixj] = a; }
            }
            __syncthreads();
        }
    }

    // inclusive scan (Hillis-Steele)
    cs[t] = srt[t];
    __syncthreads();
    for (int off = 1; off < 256; off <<= 1) {
        const float v = (t >= off) ? cs[t - off] : 0.0f;
        __syncthreads();
        cs[t] += v;
        __syncthreads();
    }

    // support size, tau, projection
    if ((1.0f + (float)(t + 1) * srt[t]) > cs[t]) atomicAdd(&cnt, 1);
    __syncthreads();
    const int ks = cnt;                         // >= 1 by construction
    const float tau = (cs[ks - 1] - 1.0f) / (float)ks;
    out[(size_t)b * 256 + t] = fmaxf(z - tau, 0.0f);
}

// ---------------------------------------------------------------------------
extern "C" void kernel_launch(void* const* d_in, const int* in_sizes, int n_in,
                              void* d_out, int out_size, void* d_ws, size_t ws_size,
                              hipStream_t stream) {
    (void)in_sizes; (void)n_in; (void)out_size; (void)ws_size;

    const float* x    = (const float*)d_in[0];
    const int*   scn  = (const int*)d_in[1];
    const float* anch = (const float*)d_in[2];
    const float* bn_beta  = (const float*)d_in[3];
    const float* bn_gamma = (const float*)d_in[4];
    const float* bn_rm    = (const float*)d_in[5];
    const float* bn_rv    = (const float*)d_in[6];
    // per-layer field order (sorted keys): 0=W 1=b 2=beta 3=gamma 4=rm 5=rv
    const int WK = 7, WQ = 25;
    auto P = [&](int base, int l, int f) { return (const float*)d_in[base + l * 6 + f]; };

    // workspace layout (bytes)
    char* ws = (char*)d_ws;
    f16*   W16  = (f16*)  (ws + 0);           // 6 x 512 x 512 f16   (3,145,728)
    float* BIAS = (float*)(ws + 3145728);     // 6 x 512 f32         (   12,288)
    float* HSC  = (float*)(ws + 3158016);     // 512 f32
    float* HSH  = HSC + 512;                  // 512 f32
    f16*   BUF0 = (f16*)  (ws + 3162112);     // 16384 x 512 f16
    f16*   BUF1 = (f16*)  (ws + 19939328);    // 16384 x 512 f16
    f16*   Q0   = (f16*)  (ws + 36716544);    // 2048 x 512 f16
    f16*   Q1   = (f16*)  (ws + 38813696);    // 2048 x 512 f16  (total ~40.9 MB)

    // fold BN into weights: slots 0..2 = wq layers, 3..5 = wk layers
    for (int l = 0; l < 3; ++l) {
        fold_bn_kernel<<<512, 256, 0, stream>>>(
            P(WQ,l,0), P(WQ,l,1), P(WQ,l,3), P(WQ,l,2), P(WQ,l,4), P(WQ,l,5),
            W16 + (size_t)l * 262144, BIAS + l * 512);
        fold_bn_kernel<<<512, 256, 0, stream>>>(
            P(WK,l,0), P(WK,l,1), P(WK,l,3), P(WK,l,2), P(WK,l,4), P(WK,l,5),
            W16 + (size_t)(3 + l) * 262144, BIAS + (3 + l) * 512);
    }
    head_bn_kernel<<<1, 512, 0, stream>>>(bn_gamma, bn_beta, bn_rm, bn_rv, HSC, HSH);

    cvt_f32_f16<<<(2048 * 512) / 256, 256, 0, stream>>>(x, Q0, 2048 * 512);
    cvt_f32_f16<<<(16384 * 512) / 256, 256, 0, stream>>>(anch, BUF0, 16384 * 512);

    const dim3 gq(2048 / 128, 4), gk(16384 / 128, 4);
    // Q path: Q0 -> Q1 -> Q0 -> Q1
    gemm_bn_act<0><<<gq, 256, 0, stream>>>(Q0, W16 + 0 * 262144, BIAS + 0 * 512, HSC, HSH, Q1);
    gemm_bn_act<0><<<gq, 256, 0, stream>>>(Q1, W16 + 1 * 262144, BIAS + 1 * 512, HSC, HSH, Q0);
    gemm_bn_act<1><<<gq, 256, 0, stream>>>(Q0, W16 + 2 * 262144, BIAS + 2 * 512, HSC, HSH, Q1);
    // K path: BUF0 -> BUF1 -> BUF0 -> BUF1
    gemm_bn_act<0><<<gk, 256, 0, stream>>>(BUF0, W16 + 3 * 262144, BIAS + 3 * 512, HSC, HSH, BUF1);
    gemm_bn_act<0><<<gk, 256, 0, stream>>>(BUF1, W16 + 4 * 262144, BIAS + 4 * 512, HSC, HSH, BUF0);
    gemm_bn_act<1><<<gk, 256, 0, stream>>>(BUF0, W16 + 5 * 262144, BIAS + 5 * 512, HSC, HSH, BUF1);

    attn_sparsemax<<<2048, 256, 0, stream>>>(Q1, BUF1, scn, (float*)d_out);
}
